// RWKV_TimeMix_12369505812813
// MI455X (gfx1250) — matrix operand order repaired and checked by reading, not compiled, not run on hardware
//
#include <hip/hip_runtime.h>

// ---------------------------------------------------------------------------
// RWKV TimeMix forward for MI455X (gfx1250, wave32, WMMA bf16)
// All matmul stages on v_wmma_f32_16x16x32_bf16; fp32->bf16 conversion hoisted
// into one-time elementwise pre-passes; time_w window staged to LDS via the
// CDNA5 async global->LDS path (ASYNCcnt).
// ---------------------------------------------------------------------------

typedef __attribute__((ext_vector_type(16))) __bf16 v16bf;
typedef __attribute__((ext_vector_type(8)))  __bf16 v8bf;
typedef __attribute__((ext_vector_type(4)))  __bf16 v4bf;
typedef __attribute__((ext_vector_type(8)))  float  v8f;

constexpr int kB      = 4;
constexpr int kT      = 2048;
constexpr int kC      = 1024;
constexpr int kH      = 16;
constexpr int kA      = 1024;
constexpr int kHS     = 64;
constexpr int kTT     = 2048;
constexpr int kHalf   = kC / 2;
constexpr int kChunks = 16;            // cumsum chunks along T
constexpr int kChunkT = kT / kChunks;  // 128

// ---- fragment load helpers -------------------------------------------------
// A fragment (16x32 bf16): lane holds row M=lane%16, K = (lane/16)*8 + {0..7}
// then +16 (8-interleave).  base points at element (row, k0 + (lane>>4)*8).
__device__ __forceinline__ v16bf load_a_frag(const __bf16* base) {
  v8bf lo = *(const v8bf*)(base);
  v8bf hi = *(const v8bf*)(base + 16);
  return __builtin_shufflevector(lo, hi, 0, 1, 2, 3, 4, 5, 6, 7,
                                 8, 9, 10, 11, 12, 13, 14, 15);
}
// B fragment (32x16 bf16): lane holds col N=lane%16, K = (lane/16)*8 + {0..7}
// then +16 (8-interleave).  base points at element (col, k0 + (lane>>4)*8).
__device__ __forceinline__ v16bf load_b_frag(const __bf16* base) {
  v8bf lo = *(const v8bf*)(base);
  v8bf hi = *(const v8bf*)(base + 16);
  return __builtin_shufflevector(lo, hi, 0, 1, 2, 3, 4, 5, 6, 7,
                                 8, 9, 10, 11, 12, 13, 14, 15);
}

// ---------------------------------------------------------------------------
// Prepass A: generic fp32 -> bf16 conversion (weights, time_w), 4 elems/thread.
// ---------------------------------------------------------------------------
__global__ __launch_bounds__(256)
void cvt_f32_bf16(const float* __restrict__ src, __bf16* __restrict__ dst, int n)
{
  const int i = (blockIdx.x * 256 + threadIdx.x) * 4;
  if (i >= n) return;
  const float4 f = *(const float4*)(src + i);
  v4bf o;
  o[0] = (__bf16)f.x; o[1] = (__bf16)f.y; o[2] = (__bf16)f.z; o[3] = (__bf16)f.w;
  *(v4bf*)(dst + i) = o;
}

// ---------------------------------------------------------------------------
// Prepass B: time-shift x (first C/2 channels delayed one step) -> bf16.
// ---------------------------------------------------------------------------
__global__ __launch_bounds__(256)
void xshift_cvt(const float* __restrict__ x, __bf16* __restrict__ xs)
{
  const size_t i = ((size_t)blockIdx.x * 256 + threadIdx.x) * 4;
  const int c = (int)(i % kC);
  const int t = (int)((i / kC) % kT);
  const int b = (int)(i / ((size_t)kC * kT));
  v4bf o;
  if (c < kHalf && t == 0) {
    o[0] = o[1] = o[2] = o[3] = (__bf16)0.0f;
  } else {
    const int st = (c < kHalf) ? (t - 1) : t;
    const float4 f = *(const float4*)(x + ((size_t)(b * kT + st) * kC + c));
    o[0] = (__bf16)f.x; o[1] = (__bf16)f.y; o[2] = (__bf16)f.z; o[3] = (__bf16)f.w;
  }
  *(v4bf*)(xs + i) = o;
}

// ---------------------------------------------------------------------------
// Kernel 1: fused k/v/r GEMM on bf16 operands.
//   k = exp(clip(xs @ Wk^T + bk)), v = xs @ Wv^T + bv, r -> sigmoid(...).
// Block (32,8): each wave owns two 16x16 output tiles (shared B fragments).
// Grid (A/16, (B*T)/256).  6 WMMA per K-step per wave.
// ---------------------------------------------------------------------------
__global__ __launch_bounds__(256)
void kvr_gemm(const __bf16* __restrict__ xs,
              const __bf16* __restrict__ Wkb, const float* __restrict__ bk,
              const __bf16* __restrict__ Wvb, const float* __restrict__ bv,
              const __bf16* __restrict__ Wrb, const float* __restrict__ br,
              __bf16* __restrict__ kq, __bf16* __restrict__ vq,
              __bf16* __restrict__ rq)
{
  const int lane  = threadIdx.x;
  const int wave  = threadIdx.y;
  const int n     = blockIdx.x * 16 + (lane & 15);
  const int m0    = blockIdx.y * 256 + wave * 32;
  const int kofsA = (lane >> 4) * 8;
  const int kofsB = (lane >> 4) * 8;
  const size_t ra0 = (size_t)(m0 + (lane & 15)) * kC + kofsA;
  const size_t ra1 = ra0 + (size_t)16 * kC;
  const size_t rb  = (size_t)n * kC + kofsB;

  v8f ck0 = {}, cv0 = {}, cr0 = {};
  v8f ck1 = {}, cv1 = {}, cr1 = {};

  for (int k0 = 0; k0 < kC; k0 += 32) {
    if (k0 + 32 < kC) {
      __builtin_prefetch(Wkb + rb + k0 + 32, 0, 3);
      __builtin_prefetch(Wvb + rb + k0 + 32, 0, 3);
      __builtin_prefetch(Wrb + rb + k0 + 32, 0, 3);
    }
    const v16bf a0 = load_a_frag(xs + ra0 + k0);
    const v16bf a1 = load_a_frag(xs + ra1 + k0);
    const v16bf fk = load_b_frag(Wkb + rb + k0);
    const v16bf fv = load_b_frag(Wvb + rb + k0);
    const v16bf fr = load_b_frag(Wrb + rb + k0);
    ck0 = __builtin_amdgcn_wmma_f32_16x16x32_bf16(false, a0, false, fk, (short)0, ck0, false, false);
    cv0 = __builtin_amdgcn_wmma_f32_16x16x32_bf16(false, a0, false, fv, (short)0, cv0, false, false);
    cr0 = __builtin_amdgcn_wmma_f32_16x16x32_bf16(false, a0, false, fr, (short)0, cr0, false, false);
    ck1 = __builtin_amdgcn_wmma_f32_16x16x32_bf16(false, a1, false, fk, (short)0, ck1, false, false);
    cv1 = __builtin_amdgcn_wmma_f32_16x16x32_bf16(false, a1, false, fv, (short)0, cv1, false, false);
    cr1 = __builtin_amdgcn_wmma_f32_16x16x32_bf16(false, a1, false, fr, (short)0, cr1, false, false);
  }

  const float bkn = bk[n], bvn = bv[n], brn = br[n];
  #pragma unroll
  for (int tile = 0; tile < 2; ++tile) {
    const v8f& ck = tile ? ck1 : ck0;
    const v8f& cv = tile ? cv1 : cv0;
    const v8f& cr = tile ? cr1 : cr0;
    #pragma unroll
    for (int j = 0; j < 8; ++j) {
      const int row = m0 + tile * 16 + j + kofsA;   // M = j + 8*(lane/16)
      const size_t g = (size_t)row * kA + n;
      float kk = ck[j] + bkn;
      kk = fminf(fmaxf(kk, -60.0f), 30.0f);
      kq[g] = (__bf16)__expf(kk);
      vq[g] = (__bf16)(cv[j] + bvn);
      rq[g] = (__bf16)(1.0f / (1.0f + __expf(-(cr[j] + brn))));
    }
  }
}

// ---------------------------------------------------------------------------
// Kernel 2 (3-phase blocked scan): sum_k = cumsum(k, axis=T)
// Phase 1: per-chunk channel sums.  Phase 2: exclusive prefix over chunks.
// Phase 3: re-accumulate within chunk starting from the chunk offset.
// ---------------------------------------------------------------------------
__global__ __launch_bounds__(256)
void scan_partial(const __bf16* __restrict__ kq, float* __restrict__ chunkTot)
{
  const int b  = blockIdx.x / kChunks;
  const int ch = blockIdx.x % kChunks;
  const int c4 = threadIdx.x * 4;
  float acc[4] = {0.f, 0.f, 0.f, 0.f};
  size_t g = ((size_t)b * kT + ch * kChunkT) * kA + c4;
  for (int tt = 0; tt < kChunkT; ++tt, g += kA) {
    const v4bf d = *(const v4bf*)(kq + g);
    #pragma unroll
    for (int j = 0; j < 4; ++j) acc[j] += (float)d[j];
  }
  float4 o = {acc[0], acc[1], acc[2], acc[3]};
  *(float4*)(chunkTot + ((size_t)(b * kChunks + ch)) * kA + c4) = o;
}

__global__ __launch_bounds__(256)
void scan_prefix(float* __restrict__ chunkTot)
{
  const int b = blockIdx.x;
  const int a = blockIdx.y * 256 + threadIdx.x;
  float run = 0.0f;
  for (int j = 0; j < kChunks; ++j) {
    const size_t idx = ((size_t)(b * kChunks + j)) * kA + a;
    const float t = chunkTot[idx];
    chunkTot[idx] = run;                // exclusive prefix
    run += t;
  }
}

__global__ __launch_bounds__(256)
void scan_apply(const __bf16* __restrict__ kq,
                const float* __restrict__ chunkTot, float* __restrict__ sumk)
{
  const int b  = blockIdx.x / kChunks;
  const int ch = blockIdx.x % kChunks;
  const int c4 = threadIdx.x * 4;
  const float4 base = *(const float4*)(chunkTot + ((size_t)(b * kChunks + ch)) * kA + c4);
  float acc[4] = {base.x, base.y, base.z, base.w};
  size_t g = ((size_t)b * kT + ch * kChunkT) * kA + c4;
  for (int tt = 0; tt < kChunkT; ++tt, g += kA) {
    const v4bf d = *(const v4bf*)(kq + g);
    #pragma unroll
    for (int j = 0; j < 4; ++j) acc[j] += (float)d[j];
    float4 o = {acc[0], acc[1], acc[2], acc[3]};
    *(float4*)(sumk + g) = o;
  }
}

// ---------------------------------------------------------------------------
// Prepass C: kv = k * v * alpha[h,u]   (bf16), 4 elems/thread.
// ---------------------------------------------------------------------------
__global__ __launch_bounds__(256)
void kv_mul(const __bf16* __restrict__ kq, const __bf16* __restrict__ vq,
            const float* __restrict__ alpha, __bf16* __restrict__ kvq)
{
  const size_t i = ((size_t)blockIdx.x * 256 + threadIdx.x) * 4;
  const int a = (int)(i % kA);
  const int u = (int)((i / kA) % kT);
  const int h = a / kHS;
  const float al = alpha[(size_t)h * kTT + u];
  const v4bf kk = *(const v4bf*)(kq + i);
  const v4bf vv = *(const v4bf*)(vq + i);
  v4bf o;
  #pragma unroll
  for (int j = 0; j < 4; ++j)
    o[j] = (__bf16)((float)kk[j] * (float)vv[j] * al);
  *(v4bf*)(kvq + i) = o;
}

// ---------------------------------------------------------------------------
// Kernel 3: causal Toeplitz matmul per (b,h):
//   wkv[t,c] = sum_{u<=t} time_w[h, TT-1+u-t] * kv[u,c]   (alpha pre-folded)
// then rwkv = sigmoid(r) * wkv * beta[h,t] / sum_k  (bf16).
// kv K-tile staged to LDS transposed [c][u] (stride 40 -> 16B-aligned cols);
// time_w window (95 contiguous bf16) staged via async global->LDS copy.
// Block (32,8) covers 64(t) x 64(c); grid (B*H, T/64).
// ---------------------------------------------------------------------------
__global__ __launch_bounds__(256)
void wkv_kernel(const __bf16* __restrict__ kvq,
                const __bf16* __restrict__ rq,
                const float* __restrict__ sumk,
                const __bf16* __restrict__ twb,
                const float* __restrict__ beta,
                __bf16* __restrict__ rwkv)
{
  constexpr int kLdsStride = 40;                  // 80B rows: 16B-aligned cols
  __shared__ __align__(16) __bf16 kvt[64 * kLdsStride];
  __shared__ __align__(16) __bf16 wseg[128];      // time_w window for this K-step

  const int lane  = threadIdx.x;
  const int wave  = threadIdx.y;
  const int tid   = wave * 32 + lane;
  const int b     = blockIdx.x / kH;
  const int h     = blockIdx.x % kH;
  const int t0    = blockIdx.y * 64;
  const int mi    = wave >> 1;                    // row subtile 0..3
  const int ni0   = (wave & 1) * 2;               // col subtiles {0,1} or {2,3}
  const int kofsA = (lane >> 4) * 8;
  const int kofsB = (lane >> 4) * 8;
  const int trow  = t0 + mi * 16 + (lane & 15);
  const int lofs  = t0 + 63 - trow;               // per-lane window offset
  const int tmax  = t0 + mi * 16 + 15;            // last row this wave owns
  const int kend  = t0 + 64;
  const __bf16* wrow = twb + (size_t)h * kTT;

  v8f acc0 = {}, acc1 = {};

  for (int u0 = 0; u0 < kend; u0 += 32) {
    // ---- async stage: time_w window wrow[base .. base+127] -> LDS ----------
    // wseg[j] == time_w[h, TT-1+u-t] for j = (u-u0) + (t0+63-t); 95 values
    // used, tail over-read is workspace-internal and masked below.
    if (tid < 16) {
      const int base = u0 - t0 + (kTT - 1 - 63);  // multiple of 32
      const unsigned ldsa = (unsigned)(uintptr_t)(&wseg[tid * 8]);
      const unsigned long long ga =
          (unsigned long long)(uintptr_t)(wrow + base + tid * 8);
      asm volatile("global_load_async_to_lds_b128 %0, %1, off"
                   :: "v"(ldsa), "v"(ga) : "memory");
    }
    // ---- cooperative stage of kv[u0..u0+31][0..63] into LDS (transposed) ---
    {
      const int li = tid * 8;                     // 256 thr * 8 = 2048 elems
      const int ul = li >> 6;                     // local u row
      const int c  = li & 63;
      const v8bf d = *(const v8bf*)(kvq + ((size_t)(b * kT + u0 + ul)) * kA
                                        + h * kHS + c);
      #pragma unroll
      for (int j = 0; j < 8; ++j)
        kvt[(c + j) * kLdsStride + ul] = d[j];
    }
    asm volatile("s_wait_asynccnt 0x0" ::: "memory");
    __syncthreads();

    if (u0 <= tmax) {                             // wave-uniform causal skip
      // ---- A fragment: decay tile from LDS window, uniform-threshold mask --
      const int thr = t0 + 63 - u0;               // block-uniform
      v16bf af;
      #pragma unroll
      for (int r = 0; r < 2; ++r) {
        #pragma unroll
        for (int i = 0; i < 8; ++i) {
          const int idx = kofsA + r * 16 + i + lofs;
          af[r * 8 + i] = (idx <= thr) ? wseg[idx] : (__bf16)0.0f;
        }
      }
      // ---- B fragments: 16B vector loads from transposed LDS ----
      const int c0 = ni0 * 16 + (lane & 15);
      const v16bf b0 = load_b_frag(kvt + c0 * kLdsStride + kofsB);
      const v16bf b1 = load_b_frag(kvt + (c0 + 16) * kLdsStride + kofsB);
      acc0 = __builtin_amdgcn_wmma_f32_16x16x32_bf16(false, af, false, b0, (short)0, acc0, false, false);
      acc1 = __builtin_amdgcn_wmma_f32_16x16x32_bf16(false, af, false, b1, (short)0, acc1, false, false);
    }
    __syncthreads();
  }

  // ---- epilogue: *beta, *sigmoid(r), /sum_k ----
  #pragma unroll
  for (int j = 0; j < 8; ++j) {
    const int t = t0 + mi * 16 + j + kofsA;
    const float bet = beta[(size_t)h * kTT + t];
    const int c0 = ni0 * 16 + (lane & 15);
    {
      const size_t g = ((size_t)(b * kT + t)) * kA + h * kHS + c0;
      rwkv[g] = (__bf16)((float)rq[g] * (acc0[j] * bet) / sumk[g]);
    }
    {
      const size_t g = ((size_t)(b * kT + t)) * kA + h * kHS + c0 + 16;
      rwkv[g] = (__bf16)((float)rq[g] * (acc1[j] * bet) / sumk[g]);
    }
  }
}

// ---------------------------------------------------------------------------
// Kernel 4: out = (rwkv @ Wo^T + bo) * gamma[t]  (fp32 out).
// Two M-tiles per wave; grid (C/16, (B*T)/256).
// ---------------------------------------------------------------------------
__global__ __launch_bounds__(256)
void out_gemm(const __bf16* __restrict__ rwkv,
              const __bf16* __restrict__ Wob, const float* __restrict__ bo,
              const float* __restrict__ gamma, float* __restrict__ out)
{
  const int lane  = threadIdx.x;
  const int wave  = threadIdx.y;
  const int n     = blockIdx.x * 16 + (lane & 15);
  const int m0    = blockIdx.y * 256 + wave * 32;
  const int kofsA = (lane >> 4) * 8;
  const int kofsB = (lane >> 4) * 8;
  const size_t ra0 = (size_t)(m0 + (lane & 15)) * kA + kofsA;
  const size_t ra1 = ra0 + (size_t)16 * kA;
  const size_t rb  = (size_t)n * kA + kofsB;

  v8f c0 = {}, c1 = {};
  for (int k0 = 0; k0 < kA; k0 += 32) {
    if (k0 + 32 < kA) __builtin_prefetch(Wob + rb + k0 + 32, 0, 3);
    const v16bf a0 = load_a_frag(rwkv + ra0 + k0);
    const v16bf a1 = load_a_frag(rwkv + ra1 + k0);
    const v16bf bf = load_b_frag(Wob + rb + k0);
    c0 = __builtin_amdgcn_wmma_f32_16x16x32_bf16(false, a0, false, bf, (short)0, c0, false, false);
    c1 = __builtin_amdgcn_wmma_f32_16x16x32_bf16(false, a1, false, bf, (short)0, c1, false, false);
  }

  const float bon = bo[n];
  #pragma unroll
  for (int tile = 0; tile < 2; ++tile) {
    const v8f& cc = tile ? c1 : c0;
    #pragma unroll
    for (int j = 0; j < 8; ++j) {
      const int row = m0 + tile * 16 + j + kofsA;
      out[(size_t)row * kC + n] = (cc[j] + bon) * gamma[row % kT];
    }
  }
}

// ---------------------------------------------------------------------------
extern "C" void kernel_launch(void* const* d_in, const int* in_sizes, int n_in,
                              void* d_out, int out_size, void* d_ws, size_t ws_size,
                              hipStream_t stream) {
  const float* x      = (const float*)d_in[0];
  const float* time_w = (const float*)d_in[1];
  const float* alpha  = (const float*)d_in[2];
  const float* beta   = (const float*)d_in[3];
  const float* gamma  = (const float*)d_in[4];
  const float* Wk     = (const float*)d_in[5];
  const float* bk     = (const float*)d_in[6];
  const float* Wv     = (const float*)d_in[7];
  const float* bv     = (const float*)d_in[8];
  const float* Wr     = (const float*)d_in[9];
  const float* br     = (const float*)d_in[10];
  const float* Wo     = (const float*)d_in[11];
  const float* bo     = (const float*)d_in[12];
  float* out          = (float*)d_out;

  const size_t NE  = (size_t)kB * kT * kA;     // 8.39M
  const size_t NW  = (size_t)kA * kC;          // 1.05M
  const size_t NTW = (size_t)kH * kTT;         // 32K
  const size_t NCT = (size_t)kB * kChunks * kA;// 64K

  float*  sumk     = (float*)d_ws;             // fp32 first (alignment)
  float*  chunkTot = sumk + NE;
  __bf16* p    = (__bf16*)(chunkTot + NCT);
  __bf16* xs   = p; p += (size_t)kB * kT * kC;
  __bf16* Wkb  = p; p += NW;
  __bf16* Wvb  = p; p += NW;
  __bf16* Wrb  = p; p += NW;
  __bf16* Wob  = p; p += NW;
  __bf16* twb  = p; p += NTW;
  __bf16* kq   = p; p += NE;
  __bf16* vq   = p; p += NE;
  __bf16* rq   = p; p += NE;
  __bf16* kvq  = p; p += NE;
  __bf16* rwkv = p; p += NE;

  dim3 blk(32, 8);

  // prepasses: bf16 conversion
  cvt_f32_bf16<<<(int)(NW / 4 / 256), 256, 0, stream>>>(Wk, Wkb, (int)NW);
  cvt_f32_bf16<<<(int)(NW / 4 / 256), 256, 0, stream>>>(Wv, Wvb, (int)NW);
  cvt_f32_bf16<<<(int)(NW / 4 / 256), 256, 0, stream>>>(Wr, Wrb, (int)NW);
  cvt_f32_bf16<<<(int)(NW / 4 / 256), 256, 0, stream>>>(Wo, Wob, (int)NW);
  cvt_f32_bf16<<<(int)(NTW / 4 / 256), 256, 0, stream>>>(time_w, twb, (int)NTW);
  xshift_cvt<<<(int)((size_t)kB * kT * kC / 4 / 256), 256, 0, stream>>>(x, xs);

  // stage 1: k/v/r projections
  kvr_gemm<<<dim3(kA / 16, (kB * kT) / 256), blk, 0, stream>>>(
      xs, Wkb, bk, Wvb, bv, Wrb, br, kq, vq, rq);

  // stage 2: blocked cumulative sum of k, and kv = k*v*alpha
  scan_partial<<<kB * kChunks, 256, 0, stream>>>(kq, chunkTot);
  scan_prefix<<<dim3(kB, kA / 256), 256, 0, stream>>>(chunkTot);
  scan_apply<<<kB * kChunks, 256, 0, stream>>>(kq, chunkTot, sumk);
  kv_mul<<<(int)(NE / 4 / 256), 256, 0, stream>>>(kq, vq, alpha, kvq);

  // stage 3: causal decay attention
  wkv_kernel<<<dim3(kB * kH, kT / 64), blk, 0, stream>>>(
      kvq, rq, sumk, twb, beta, rwkv);

  // stage 4: output projection
  out_gemm<<<dim3(kC / 16, (kB * kT) / 256), blk, 0, stream>>>(
      rwkv, Wob, bo, gamma, out);
}
